// DecoderLSTM_noAttention_79328045957426
// MI455X (gfx1250) — matrix operand order repaired and checked by reading, not compiled, not run on hardware
//
#include <hip/hip_runtime.h>
#include <hip/hip_bf16.h>
#include <stdint.h>

// ---------------- problem constants ----------------
#define EMBED   512
#define HIDDEN  512
#define VOCAB   32000
#define B_SZ    64
#define NPIX    196
#define MAXLEN  32
#define GATES4  (4 * HIDDEN)   // 2048

typedef __attribute__((ext_vector_type(16))) _Float16 v16h;
typedef __attribute__((ext_vector_type(8)))  _Float16 v8h;
typedef __attribute__((ext_vector_type(8)))  float    v8f;

union V16U {
    v16h v;
    struct { v8h lo; v8h hi; } p;
};

// A-operand (16-bit 16x32): lane L<16 = row M=L holds CONTIGUOUS K range
// k0..k0+15 (halves 0-15); lanes 16-31 hold k0+16..k0+31.
// Caller passes p = row_base + k0 + ((lane&16)?16:0).
__device__ __forceinline__ v16h load_a16(const _Float16* p) {
    V16U u;
    u.p.lo = *(const v8h*)(p);
    u.p.hi = *(const v8h*)(p + 8);
    return u.v;
}

// B-operand (16-bit 32x16, same k order as load_a16): lane = column N,
// halves 0..15 = CONTIGUOUS K range: lanes 0-15 -> k0..k0+15, lanes 16-31 ->
// k0+16..k0+31. Caller passes p = col_base + k0 + ((lane&16)?16:0).
__device__ __forceinline__ v16h load_b16(const _Float16* p) {
    V16U u;
    u.p.lo = *(const v8h*)(p);
    u.p.hi = *(const v8h*)(p + 8);
    return u.v;
}

__device__ __forceinline__ float sigmoidf(float x) {
    return 1.0f / (1.0f + __expf(-x));
}

// ---------------- one-time prep kernels ----------------

__global__ void cvt_f32_to_f16_kernel(const float* __restrict__ src,
                                      _Float16* __restrict__ dst, int n) {
    int i = blockIdx.x * blockDim.x + threadIdx.x;
    if (i < n) dst[i] = (_Float16)src[i];
}

// summary[b][k] = mean_p enc[b][p][k]
__global__ void summary_kernel(const float* __restrict__ enc,
                               float* __restrict__ summary) {
    int i = blockIdx.x * blockDim.x + threadIdx.x;   // 64*512
    if (i >= B_SZ * HIDDEN) return;
    int b = i >> 9, k = i & (HIDDEN - 1);
    const float* p = enc + (size_t)b * NPIX * HIDDEN + k;
    float s = 0.f;
    for (int n = 0; n < NPIX; ++n) s += p[(size_t)n * HIDDEN];
    summary[i] = s * (1.0f / (float)NPIX);
}

// h = summary @ W_init_h^T + b_init_h ; c likewise (one-time, fp32 exact)
__global__ void init_hc_kernel(const float* __restrict__ summary,
                               const float* __restrict__ Wih0, const float* __restrict__ bh0,
                               const float* __restrict__ Wic0, const float* __restrict__ bc0,
                               float* __restrict__ h, float* __restrict__ c) {
    int i = blockIdx.x * blockDim.x + threadIdx.x;   // 64*512
    if (i >= B_SZ * HIDDEN) return;
    int b = i >> 9, j = i & (HIDDEN - 1);
    const float* s = summary + (size_t)b * HIDDEN;
    const float* wh = Wih0 + (size_t)j * HIDDEN;
    const float* wc = Wic0 + (size_t)j * HIDDEN;
    float ah = bh0[j], ac = bc0[j];
    for (int k = 0; k < HIDDEN; ++k) {
        float sk = s[k];
        ah = fmaf(sk, wh[k], ah);
        ac = fmaf(sk, wc[k], ac);
    }
    h[i] = ah;
    c[i] = ac;
}

__global__ void init_tok_kernel(const int* __restrict__ captions, int* __restrict__ tok) {
    int b = threadIdx.x;
    if (b < B_SZ) tok[b] = captions[(size_t)b * MAXLEN + 0];
}

// zero timestep 0 of the output (other steps are fully overwritten)
__global__ void zero_step0_kernel(float* __restrict__ out) {
    int i = blockIdx.x * blockDim.x + threadIdx.x;   // 64*32000
    if (i >= B_SZ * VOCAB) return;
    int b = i / VOCAB, v = i - b * VOCAB;
    out[(size_t)b * MAXLEN * VOCAB + v] = 0.0f;
}

// ---------------- per-step kernels ----------------

// x16 = (f16) embedding[tok[b]][:] ; h16 = (f16) h
__global__ void embed_cvt_kernel(const float* __restrict__ embedding,
                                 const int* __restrict__ tok,
                                 const float* __restrict__ h,
                                 _Float16* __restrict__ x16,
                                 _Float16* __restrict__ h16) {
    int i = blockIdx.x * blockDim.x + threadIdx.x;   // 2 * 64*512
    if (i < B_SZ * EMBED) {
        int b = i >> 9, j = i & (EMBED - 1);
        x16[i] = (_Float16)embedding[(size_t)tok[b] * EMBED + j];
    } else if (i < 2 * B_SZ * EMBED) {
        int t = i - B_SZ * EMBED;
        h16[t] = (_Float16)h[t];
    }
}

// gates[64][2048] = x16 @ Wih16^T + h16 @ Whh16^T   (raw; biases added in cell)
// One wave per 16x16 output tile: 4 M-tiles x 128 N-tiles = 512 waves
// (latency-bound small GEMM -> maximize wave parallelism).
__global__ void gates_wmma_kernel(const _Float16* __restrict__ x16,
                                  const _Float16* __restrict__ h16,
                                  const _Float16* __restrict__ Wih,
                                  const _Float16* __restrict__ Whh,
                                  float* __restrict__ gates) {
    int wave = (blockIdx.x * blockDim.x + threadIdx.x) >> 5;
    int lane = threadIdx.x & 31;
    int mt = wave & 3;        // batch tile 0..3
    int nt = wave >> 2;       // gate-col tile 0..127
    int arow = (mt << 4) + (lane & 15);      // batch row for A
    int brow = (nt << 4) + (lane & 15);      // weight row (= output col) for B
    int aksel = (lane & 16) ? 16 : 0;
    int bksel = (lane & 16) ? 16 : 0;

    const _Float16* ap = x16 + (size_t)arow * EMBED + aksel;
    const _Float16* bp = Wih + (size_t)brow * EMBED + bksel;
    v8f acc = {};
#pragma unroll
    for (int k0 = 0; k0 < EMBED; k0 += 32) {
        v16h a = load_a16(ap + k0);
        v16h b = load_b16(bp + k0);
        acc = __builtin_amdgcn_wmma_f32_16x16x32_f16(false, a, false, b,
                                                     (short)0, acc, false, false);
    }
    ap = h16 + (size_t)arow * HIDDEN + aksel;
    bp = Whh + (size_t)brow * HIDDEN + bksel;
#pragma unroll
    for (int k0 = 0; k0 < HIDDEN; k0 += 32) {
        v16h a = load_a16(ap + k0);
        v16h b = load_b16(bp + k0);
        acc = __builtin_amdgcn_wmma_f32_16x16x32_f16(false, a, false, b,
                                                     (short)0, acc, false, false);
    }
    // D layout: lane n (0-15) VGPR r -> (M=r, N=n); lane n+16 VGPR r -> (M=r+8, N=n)
    int n = (nt << 4) + (lane & 15);
    int mbase = (mt << 4) + ((lane & 16) ? 8 : 0);
#pragma unroll
    for (int r = 0; r < 8; ++r)
        gates[(size_t)(mbase + r) * GATES4 + n] = acc[r];
}

// LSTM cell elementwise: reads gates + biases, updates h, c
__global__ void cell_kernel(const float* __restrict__ gates,
                            const float* __restrict__ b_ih,
                            const float* __restrict__ b_hh,
                            float* __restrict__ h, float* __restrict__ c) {
    int i = blockIdx.x * blockDim.x + threadIdx.x;   // 64*512
    if (i >= B_SZ * HIDDEN) return;
    int b = i >> 9, j = i & (HIDDEN - 1);
    const float* g = gates + (size_t)b * GATES4;
    float gi = g[j]               + b_ih[j]               + b_hh[j];
    float gf = g[HIDDEN + j]      + b_ih[HIDDEN + j]      + b_hh[HIDDEN + j];
    float gg = g[2 * HIDDEN + j]  + b_ih[2 * HIDDEN + j]  + b_hh[2 * HIDDEN + j];
    float go = g[3 * HIDDEN + j]  + b_ih[3 * HIDDEN + j]  + b_hh[3 * HIDDEN + j];
    float ii = sigmoidf(gi), ff = sigmoidf(gf), oo = sigmoidf(go);
    float gt = tanhf(gg);
    float c2 = ff * c[i] + ii * gt;
    float h2 = oo * tanhf(c2);
    c[i] = c2;
    h[i] = h2;
}

// out[b][t][v] = h16 @ Wfc16^T + b_fc
// One wave per VOCAB tile (2000 waves); each wave computes all 4 batch tiles,
// holding the B fragment in registers across 4 WMMAs (B read once per step:
// 32.8 MB from L2 instead of 131 MB).
__global__ void logits_wmma_kernel(const _Float16* __restrict__ h16,
                                   const _Float16* __restrict__ Wfc,
                                   const float* __restrict__ b_fc,
                                   float* __restrict__ out, int t) {
    int nt   = (blockIdx.x * blockDim.x + threadIdx.x) >> 5;  // 0..1999
    int lane = threadIdx.x & 31;
    int brow = (nt << 4) + (lane & 15);
    int aksel = (lane & 16) ? 16 : 0;
    int bksel = (lane & 16) ? 16 : 0;

    const _Float16* ap = h16 + (size_t)(lane & 15) * HIDDEN + aksel; // mt=0 rows
    const _Float16* bp = Wfc + (size_t)brow * HIDDEN + bksel;

    v8f acc0 = {}, acc1 = {}, acc2 = {}, acc3 = {};
#pragma unroll
    for (int k0 = 0; k0 < HIDDEN; k0 += 32) {
        v16h b  = load_b16(bp + k0);
        v16h a0 = load_a16(ap + k0);
        v16h a1 = load_a16(ap + 16 * HIDDEN + k0);
        v16h a2 = load_a16(ap + 32 * HIDDEN + k0);
        v16h a3 = load_a16(ap + 48 * HIDDEN + k0);
        acc0 = __builtin_amdgcn_wmma_f32_16x16x32_f16(false, a0, false, b,
                                                      (short)0, acc0, false, false);
        acc1 = __builtin_amdgcn_wmma_f32_16x16x32_f16(false, a1, false, b,
                                                      (short)0, acc1, false, false);
        acc2 = __builtin_amdgcn_wmma_f32_16x16x32_f16(false, a2, false, b,
                                                      (short)0, acc2, false, false);
        acc3 = __builtin_amdgcn_wmma_f32_16x16x32_f16(false, a3, false, b,
                                                      (short)0, acc3, false, false);
    }
    int n = (nt << 4) + (lane & 15);
    float bias = b_fc[n];
    int mb = (lane & 16) ? 8 : 0;
    v8f accs[4] = {acc0, acc1, acc2, acc3};
#pragma unroll
    for (int mt = 0; mt < 4; ++mt) {
#pragma unroll
        for (int r = 0; r < 8; ++r) {
            int b = (mt << 4) + mb + r;  // batch row
            out[((size_t)b * MAXLEN + t) * VOCAB + n] = accs[mt][r] + bias;
        }
    }
}

// tok[b] = argmax_v out[b][t][v]  (first occurrence on ties, like jnp.argmax)
__global__ void argmax_kernel(const float* __restrict__ out, int t,
                              int* __restrict__ tok) {
    __shared__ float sval[256];
    __shared__ int   sidx[256];
    int b = blockIdx.x;
    const float* row = out + ((size_t)b * MAXLEN + t) * VOCAB;
    float best = -__builtin_inff();
    int bidx = 0;
    for (int v = threadIdx.x; v < VOCAB; v += 256) {
        float val = row[v];
        if (val > best) { best = val; bidx = v; }   // ascending v => first max kept
    }
    sval[threadIdx.x] = best;
    sidx[threadIdx.x] = bidx;
    __syncthreads();
    for (int s = 128; s > 0; s >>= 1) {
        if (threadIdx.x < s) {
            float ov = sval[threadIdx.x + s];
            int   oi = sidx[threadIdx.x + s];
            if (ov > sval[threadIdx.x] ||
                (ov == sval[threadIdx.x] && oi < sidx[threadIdx.x])) {
                sval[threadIdx.x] = ov;
                sidx[threadIdx.x] = oi;
            }
        }
        __syncthreads();
    }
    if (threadIdx.x == 0) tok[b] = sidx[0];
}

// ---------------- workspace layout (bytes) ----------------
static constexpr size_t OFF_WIH16 = 0;                                   // 2048*512*2
static constexpr size_t OFF_WHH16 = OFF_WIH16 + (size_t)GATES4 * EMBED * 2;
static constexpr size_t OFF_WFC16 = OFF_WHH16 + (size_t)GATES4 * HIDDEN * 2;
static constexpr size_t OFF_H     = OFF_WFC16 + (size_t)VOCAB * HIDDEN * 2;
static constexpr size_t OFF_C     = OFF_H     + (size_t)B_SZ * HIDDEN * 4;
static constexpr size_t OFF_X16   = OFF_C     + (size_t)B_SZ * HIDDEN * 4;
static constexpr size_t OFF_H16   = OFF_X16   + (size_t)B_SZ * EMBED * 2;
static constexpr size_t OFF_GATES = OFF_H16   + (size_t)B_SZ * HIDDEN * 2;
static constexpr size_t OFF_SUM   = OFF_GATES + (size_t)B_SZ * GATES4 * 4;
static constexpr size_t OFF_TOK   = OFF_SUM   + (size_t)B_SZ * HIDDEN * 4;

extern "C" void kernel_launch(void* const* d_in, const int* in_sizes, int n_in,
                              void* d_out, int out_size, void* d_ws, size_t ws_size,
                              hipStream_t stream) {
    (void)in_sizes; (void)n_in; (void)out_size; (void)ws_size;

    const float* enc       = (const float*)d_in[0];
    const int*   captions  = (const int*)  d_in[1];
    const float* embedding = (const float*)d_in[2];
    const float* W_ih      = (const float*)d_in[3];
    const float* b_ih      = (const float*)d_in[4];
    const float* W_hh      = (const float*)d_in[5];
    const float* b_hh      = (const float*)d_in[6];
    const float* W_fc      = (const float*)d_in[7];
    const float* b_fc      = (const float*)d_in[8];
    const float* W_init_h  = (const float*)d_in[9];
    const float* b_init_h  = (const float*)d_in[10];
    const float* W_init_c  = (const float*)d_in[11];
    const float* b_init_c  = (const float*)d_in[12];

    char* ws = (char*)d_ws;
    _Float16* Wih16 = (_Float16*)(ws + OFF_WIH16);
    _Float16* Whh16 = (_Float16*)(ws + OFF_WHH16);
    _Float16* Wfc16 = (_Float16*)(ws + OFF_WFC16);
    float*    h     = (float*)   (ws + OFF_H);
    float*    c     = (float*)   (ws + OFF_C);
    _Float16* x16   = (_Float16*)(ws + OFF_X16);
    _Float16* h16   = (_Float16*)(ws + OFF_H16);
    float*    gates = (float*)   (ws + OFF_GATES);
    float*    summ  = (float*)   (ws + OFF_SUM);
    int*      tok   = (int*)     (ws + OFF_TOK);
    float*    out   = (float*)   d_out;

    // ---- one-time prep ----
    {
        int n1 = GATES4 * EMBED;          // 1,048,576
        cvt_f32_to_f16_kernel<<<(n1 + 255) / 256, 256, 0, stream>>>(W_ih, Wih16, n1);
        cvt_f32_to_f16_kernel<<<(n1 + 255) / 256, 256, 0, stream>>>(W_hh, Whh16, n1);
        int n2 = VOCAB * HIDDEN;          // 16,384,000
        cvt_f32_to_f16_kernel<<<(n2 + 255) / 256, 256, 0, stream>>>(W_fc, Wfc16, n2);
    }
    summary_kernel<<<(B_SZ * HIDDEN + 255) / 256, 256, 0, stream>>>(enc, summ);
    init_hc_kernel<<<(B_SZ * HIDDEN + 255) / 256, 256, 0, stream>>>(
        summ, W_init_h, b_init_h, W_init_c, b_init_c, h, c);
    init_tok_kernel<<<1, 64, 0, stream>>>(captions, tok);
    zero_step0_kernel<<<(B_SZ * VOCAB + 255) / 256, 256, 0, stream>>>(out);

    // ---- sequential decode: 31 steps ----
    for (int t = 1; t < MAXLEN; ++t) {
        embed_cvt_kernel<<<(2 * B_SZ * EMBED + 255) / 256, 256, 0, stream>>>(
            embedding, tok, h, x16, h16);
        // 512 tiles, 8 waves/block -> 64 blocks x 256 threads
        gates_wmma_kernel<<<64, 256, 0, stream>>>(x16, h16, Wih16, Whh16, gates);
        cell_kernel<<<(B_SZ * HIDDEN + 255) / 256, 256, 0, stream>>>(
            gates, b_ih, b_hh, h, c);
        // 2000 vocab tiles, one wave each (4 batch tiles per wave):
        // 250 blocks x 256 threads (8 waves)
        logits_wmma_kernel<<<250, 256, 0, stream>>>(h16, Wfc16, b_fc, out, t);
        argmax_kernel<<<B_SZ, 256, 0, stream>>>(out, t, tok);
    }
}
